// CausalSelfAttention_20899310863133
// MI455X (gfx1250) — matrix operand order repaired and checked by reading, not compiled, not run on hardware
//
#include <hip/hip_runtime.h>
#include <hip/hip_bf16.h>

// CDNA5 (gfx1250) causal self-attention: bf16 WMMA GEMMs + flash attention.
// B=4, T=2048, C=1024, H=16, D=64.
// GEMM tile fills use GLOBAL_LOAD_ASYNC_TO_LDS_B128 (ASYNCcnt DMA path).

typedef __attribute__((ext_vector_type(16))) __bf16 v16bf;
typedef __attribute__((ext_vector_type(8)))  float  v8f;

union Frag {
    unsigned int u[8];
    v16bf v;
};

#define T_SEQ 2048
#define C_DIM 1024
#define H_NUM 16
#define D_HEAD 64
#define B_BATCH 4

// ---------------------------------------------------------------------------
// Kernel 1: f32 -> bf16 conversion
// ---------------------------------------------------------------------------
__global__ __launch_bounds__(256) void cvt_f32_bf16(const float* __restrict__ src,
                                                    __hip_bfloat16* __restrict__ dst,
                                                    int n) {
    int i = blockIdx.x * blockDim.x + threadIdx.x;
    if (i < n) dst[i] = __float2bfloat16(src[i]);
}

// ---------------------------------------------------------------------------
// Kernel 2/4: tiled GEMM  out[m,n] = sum_k X[m,k]*W[n,k] + bias[n]
// Block = 256 threads (8 waves), tile 128x128, K-step 32, double-buffered LDS
// filled by async-to-LDS DMA. which: 0=q ([B,H,T,D], scaled 1/8),
// 1=k ([B,H,T,D]), 2=v transposed ([B,H,D,T]), 3=proj (f32 to d_out).
// ---------------------------------------------------------------------------
__global__ __launch_bounds__(256) void gemm_bf16_wmma(
    const __hip_bfloat16* __restrict__ X,      // [8192 x 1024] bf16
    const __hip_bfloat16* __restrict__ Wbase,  // [1024 x 1024] bf16 (x3 if qkv)
    const float* __restrict__ bias_q,
    const float* __restrict__ bias_k,
    const float* __restrict__ bias_v,
    __hip_bfloat16* __restrict__ out_q,
    __hip_bfloat16* __restrict__ out_k,
    __hip_bfloat16* __restrict__ out_vt,
    float* __restrict__ out_f,
    int mode_base) {
    const int tid  = threadIdx.x;
    const int lane = tid & 31;
    const int wv   = tid >> 5;
    const int l16  = lane & 15;
    const int lh   = lane >> 4;
    const int wm   = wv & 3;   // 4 wave tiles along M (32 rows each)
    const int wn   = wv >> 2;  // 2 wave tiles along N (64 cols each)

    const int which = mode_base + (int)blockIdx.z;
    const __hip_bfloat16* W =
        Wbase + ((mode_base == 0) ? ((size_t)blockIdx.z << 20) : (size_t)0);
    const float* bias = (which == 0 || which == 3) ? bias_q
                        : (which == 1)             ? bias_k
                                                   : bias_v;

    __shared__ __align__(16) __hip_bfloat16 As[2][128 * 32];
    __shared__ __align__(16) __hip_bfloat16 Bs[2][128 * 32];

    const int ldr  = tid >> 1;        // 0..127: tile row
    const int koff = (tid & 1) * 16;  // 16 bf16 per half-row
    const size_t xrow = (size_t)(blockIdx.x * 128 + ldr) * C_DIM;
    const size_t wrow = (size_t)(blockIdx.y * 128 + ldr) * C_DIM;

    // Issue async DMA: 32B of A tile and 32B of B tile per thread.
    // INST_OFFSET applies to both the LDS and global addresses, so one
    // LDS-addr VGPR + one 64-bit global address covers both 16B chunks.
    auto loadTilesAsync = [&](int buf, int k0) {
        unsigned ldsA = (unsigned)(uintptr_t)&As[buf][ldr * 32 + koff];
        unsigned ldsB = (unsigned)(uintptr_t)&Bs[buf][ldr * 32 + koff];
        unsigned long long ga = (unsigned long long)(uintptr_t)(X + xrow + k0 + koff);
        unsigned long long gb = (unsigned long long)(uintptr_t)(W + wrow + k0 + koff);
        asm volatile(
            "global_load_async_to_lds_b128 %0, %2, off\n\t"
            "global_load_async_to_lds_b128 %0, %2, off offset:16\n\t"
            "global_load_async_to_lds_b128 %1, %3, off\n\t"
            "global_load_async_to_lds_b128 %1, %3, off offset:16"
            :
            : "v"(ldsA), "v"(ldsB), "v"(ga), "v"(gb)
            : "memory");
    };

    v8f acc[2][4];
#pragma unroll
    for (int i = 0; i < 2; i++)
#pragma unroll
        for (int j = 0; j < 4; j++) acc[i][j] = (v8f)0.0f;

    loadTilesAsync(0, 0);
    asm volatile("s_wait_asynccnt 0x0" ::: "memory");
    __syncthreads();

    for (int k0 = 0; k0 < C_DIM; k0 += 32) {
        const int buf = (k0 >> 5) & 1;
        if (k0 + 32 < C_DIM) loadTilesAsync(buf ^ 1, k0 + 32);

        Frag af[2];
#pragma unroll
        for (int fr = 0; fr < 2; fr++) {
            const int m_l = wm * 32 + fr * 16 + l16;
#pragma unroll
            for (int i = 0; i < 8; i++) {
                const int kp = (i < 4) ? (2 * i + 8 * lh) : (16 + 2 * (i - 4) + 8 * lh);
                af[fr].u[i] = *(const unsigned int*)(const void*)&As[buf][m_l * 32 + kp];
            }
        }
        Frag bfr[4];
#pragma unroll
        for (int fc = 0; fc < 4; fc++) {
            const int n_l = wn * 64 + fc * 16 + l16;
#pragma unroll
            for (int i = 0; i < 8; i++) {
                const int kp = (i < 4) ? (2 * i + 8 * lh) : (16 + 2 * (i - 4) + 8 * lh);
                bfr[fc].u[i] = *(const unsigned int*)(const void*)&Bs[buf][n_l * 32 + kp];
            }
        }
#pragma unroll
        for (int fr = 0; fr < 2; fr++)
#pragma unroll
            for (int fc = 0; fc < 4; fc++)
                acc[fr][fc] = __builtin_amdgcn_wmma_f32_16x16x32_bf16(
                    false, af[fr].v, false, bfr[fc].v, (short)0, acc[fr][fc],
                    false, false);

        // Each wave waits for its own DMA before signaling the barrier, so
        // the next iteration's buffer is LDS-visible to the whole block.
        asm volatile("s_wait_asynccnt 0x0" ::: "memory");
        __syncthreads();
    }

    // Epilogue: bias + store per element.
#pragma unroll
    for (int fr = 0; fr < 2; fr++) {
#pragma unroll
        for (int fc = 0; fc < 4; fc++) {
            const int n_g = blockIdx.y * 128 + wn * 64 + fc * 16 + l16;
            const float bv_ = bias[n_g];
#pragma unroll
            for (int r = 0; r < 8; r++) {
                const int m_g = blockIdx.x * 128 + wm * 32 + fr * 16 + r + 8 * lh;
                const float val = acc[fr][fc][r] + bv_;
                if (which == 3) {
                    out_f[(size_t)m_g * C_DIM + n_g] = val;
                } else {
                    const int bb = m_g >> 11, t = m_g & (T_SEQ - 1);
                    const int hh = n_g >> 6, d = n_g & (D_HEAD - 1);
                    if (which == 0)
                        out_q[((size_t)(bb * H_NUM + hh) * T_SEQ + t) * D_HEAD + d] =
                            __float2bfloat16(val * 0.125f);  // 1/sqrt(64)
                    else if (which == 1)
                        out_k[((size_t)(bb * H_NUM + hh) * T_SEQ + t) * D_HEAD + d] =
                            __float2bfloat16(val);
                    else
                        out_vt[((size_t)(bb * H_NUM + hh) * D_HEAD + d) * T_SEQ + t] =
                            __float2bfloat16(val);
                }
            }
        }
    }
}

// ---------------------------------------------------------------------------
// Kernel 3: flash attention. One wave handles 16 query rows; key blocks of 32.
// Q pre-scaled by 1/sqrt(D). No __syncthreads (waves diverge causally).
// ---------------------------------------------------------------------------
__global__ __launch_bounds__(256) void attn_flash_wmma(
    const __hip_bfloat16* __restrict__ Q,   // [B,H,T,D]
    const __hip_bfloat16* __restrict__ Kt,  // [B,H,T,D]
    const __hip_bfloat16* __restrict__ Vt,  // [B,H,D,T]
    __hip_bfloat16* __restrict__ Y)         // [B,T,C]
{
    const int tid = threadIdx.x;
    const int w = tid >> 5, lane = tid & 31;
    const int l16 = lane & 15, lh = lane >> 4;
    const int b = blockIdx.z, h = blockIdx.y;
    const int q0 = (blockIdx.x * 8 + w) * 16;

    __shared__ __align__(16) __hip_bfloat16 Pst[8][16 * 32];

    const size_t bh = (size_t)(b * H_NUM + h);
    const __hip_bfloat16* qp = Q + bh * T_SEQ * D_HEAD;
    const __hip_bfloat16* kp = Kt + bh * T_SEQ * D_HEAD;
    const __hip_bfloat16* vp = Vt + bh * D_HEAD * T_SEQ;

    // Q A-fragments for the two K-steps of D=64
    Frag qf[2];
#pragma unroll
    for (int kk = 0; kk < 2; kk++)
#pragma unroll
        for (int i = 0; i < 8; i++) {
            const int d = kk * 32 + ((i < 4) ? (2 * i + 8 * lh) : (16 + 2 * (i - 4) + 8 * lh));
            qf[kk].u[i] = *(const unsigned int*)(const void*)&qp[(size_t)(q0 + l16) * D_HEAD + d];
        }

    v8f acc[4];
#pragma unroll
    for (int fc = 0; fc < 4; fc++) acc[fc] = (v8f)0.0f;
    float m_r[8], l_r[8];
#pragma unroll
    for (int r = 0; r < 8; r++) { m_r[r] = -1e30f; l_r[r] = 0.0f; }

    for (int s0 = 0; s0 <= q0 + 15; s0 += 32) {
        // K B-fragments: S = Q @ K^T, contraction over d
        Frag kf[2][2];
#pragma unroll
        for (int nh = 0; nh < 2; nh++) {
            const int srow = s0 + nh * 16 + l16;
#pragma unroll
            for (int kk = 0; kk < 2; kk++)
#pragma unroll
                for (int i = 0; i < 8; i++) {
                    const int d = kk * 32 + ((i < 4) ? (2 * i + 8 * lh) : (16 + 2 * (i - 4) + 8 * lh));
                    kf[nh][kk].u[i] =
                        *(const unsigned int*)(const void*)&kp[(size_t)srow * D_HEAD + d];
                }
        }
        v8f sc[2];
        sc[0] = (v8f)0.0f; sc[1] = (v8f)0.0f;
#pragma unroll
        for (int nh = 0; nh < 2; nh++)
#pragma unroll
            for (int kk = 0; kk < 2; kk++)
                sc[nh] = __builtin_amdgcn_wmma_f32_16x16x32_bf16(
                    false, qf[kk].v, false, kf[nh][kk].v, (short)0, sc[nh],
                    false, false);

        if (s0 + 31 > q0) {  // causal mask on partial blocks
#pragma unroll
            for (int nh = 0; nh < 2; nh++)
#pragma unroll
                for (int r = 0; r < 8; r++) {
                    const int col = s0 + nh * 16 + l16;
                    const int row = q0 + r + 8 * lh;
                    if (col > row) sc[nh][r] = -1e30f;
                }
        }

        // Online softmax stats: reductions across 16-lane groups (wave32)
        float alpha[8];
#pragma unroll
        for (int r = 0; r < 8; r++) {
            float mx = fmaxf(sc[0][r], sc[1][r]);
#pragma unroll
            for (int msk = 1; msk < 16; msk <<= 1)
                mx = fmaxf(mx, __shfl_xor(mx, msk, 32));
            const float nm = fmaxf(m_r[r], mx);
            alpha[r] = __expf(m_r[r] - nm);
            m_r[r] = nm;
            const float p0 = __expf(sc[0][r] - nm);
            const float p1 = __expf(sc[1][r] - nm);
            sc[0][r] = p0; sc[1][r] = p1;
            float rs = p0 + p1;
#pragma unroll
            for (int msk = 1; msk < 16; msk <<= 1) rs += __shfl_xor(rs, msk, 32);
            l_r[r] = l_r[r] * alpha[r] + rs;
        }
#pragma unroll
        for (int fc = 0; fc < 4; fc++)
#pragma unroll
            for (int r = 0; r < 8; r++) acc[fc][r] *= alpha[r];

        // Stage P (C-layout) into per-wave LDS, reload in A-fragment layout
#pragma unroll
        for (int r = 0; r < 8; r++) {
            const int row = r + 8 * lh;
            Pst[w][row * 32 + l16]      = __float2bfloat16(sc[0][r]);
            Pst[w][row * 32 + 16 + l16] = __float2bfloat16(sc[1][r]);
        }
        asm volatile("s_wait_dscnt 0x0" ::: "memory");
        Frag pf;
#pragma unroll
        for (int i = 0; i < 8; i++) {
            const int kpos = (i < 4) ? (2 * i + 8 * lh) : (16 + 2 * (i - 4) + 8 * lh);
            pf.u[i] = *(const unsigned int*)(const void*)&Pst[w][l16 * 32 + kpos];
        }

        // O += P @ V  (V transposed -> contiguous s-pairs)
#pragma unroll
        for (int fc = 0; fc < 4; fc++) {
            Frag vf;
            const int d = fc * 16 + l16;
#pragma unroll
            for (int i = 0; i < 8; i++) {
                const int sp = (i < 4) ? (2 * i + 8 * lh) : (16 + 2 * (i - 4) + 8 * lh);
                vf.u[i] = *(const unsigned int*)(const void*)&vp[(size_t)d * T_SEQ + s0 + sp];
            }
            acc[fc] = __builtin_amdgcn_wmma_f32_16x16x32_bf16(
                false, pf.v, false, vf.v, (short)0, acc[fc], false, false);
        }
    }

    // Normalize and write y as [B,T,C] bf16
#pragma unroll
    for (int r = 0; r < 8; r++) {
        const float inv = 1.0f / l_r[r];
        const int t = q0 + r + 8 * lh;
#pragma unroll
        for (int fc = 0; fc < 4; fc++) {
            Y[((size_t)b * T_SEQ + t) * C_DIM + h * D_HEAD + fc * 16 + l16] =
                __float2bfloat16(acc[fc][r] * inv);
        }
    }
}

// ---------------------------------------------------------------------------
// Launch
// ---------------------------------------------------------------------------
extern "C" void kernel_launch(void* const* d_in, const int* in_sizes, int n_in,
                              void* d_out, int out_size, void* d_ws, size_t ws_size,
                              hipStream_t stream) {
    const float* x  = (const float*)d_in[0];
    const float* Wq = (const float*)d_in[1];
    const float* bq = (const float*)d_in[2];
    const float* Wk = (const float*)d_in[3];
    const float* bk = (const float*)d_in[4];
    const float* Wv = (const float*)d_in[5];
    const float* bv = (const float*)d_in[6];
    const float* Wp = (const float*)d_in[7];
    const float* bp = (const float*)d_in[8];
    float* out = (float*)d_out;

    const size_t M = (size_t)B_BATCH * T_SEQ;  // 8192
    const size_t CC = (size_t)C_DIM * C_DIM;   // 1M

    // Workspace layout (bf16 elements), ~88 MB total
    char* ws = (char*)d_ws;
    __hip_bfloat16* xb  = (__hip_bfloat16*)(ws);                          // 16 MB
    __hip_bfloat16* Wqb = (__hip_bfloat16*)(ws + (16u << 20));            // 2 MB
    __hip_bfloat16* Wkb = (__hip_bfloat16*)(ws + (18u << 20));            // 2 MB
    __hip_bfloat16* Wvb = (__hip_bfloat16*)(ws + (20u << 20));            // 2 MB
    __hip_bfloat16* Wpb = (__hip_bfloat16*)(ws + (22u << 20));            // 2 MB
    __hip_bfloat16* qb  = (__hip_bfloat16*)(ws + (24u << 20));            // 16 MB
    __hip_bfloat16* kb  = (__hip_bfloat16*)(ws + (40u << 20));            // 16 MB
    __hip_bfloat16* vtb = (__hip_bfloat16*)(ws + (56u << 20));            // 16 MB
    __hip_bfloat16* yb  = (__hip_bfloat16*)(ws + (72u << 20));            // 16 MB

    // 1) convert inputs to bf16
    {
        const int nx = (int)(M * C_DIM);
        cvt_f32_bf16<<<nx / 256, 256, 0, stream>>>(x, xb, nx);
        const int nw = (int)CC;
        cvt_f32_bf16<<<nw / 256, 256, 0, stream>>>(Wq, Wqb, nw);
        cvt_f32_bf16<<<nw / 256, 256, 0, stream>>>(Wk, Wkb, nw);
        cvt_f32_bf16<<<nw / 256, 256, 0, stream>>>(Wv, Wvb, nw);
        cvt_f32_bf16<<<nw / 256, 256, 0, stream>>>(Wp, Wpb, nw);
    }

    // 2) QKV projections (q scaled by 1/sqrt(D)); v stored transposed
    {
        dim3 grid(M / 128, C_DIM / 128, 3);
        gemm_bf16_wmma<<<grid, 256, 0, stream>>>(xb, Wqb, bq, bk, bv,
                                                 qb, kb, vtb, nullptr, /*mode_base=*/0);
    }

    // 3) flash attention -> y [B,T,C] bf16
    {
        dim3 grid(T_SEQ / (16 * 8), H_NUM, B_BATCH);
        attn_flash_wmma<<<grid, 256, 0, stream>>>(qb, kb, vtb, yb);
    }

    // 4) output projection -> f32 d_out
    {
        dim3 grid(M / 128, C_DIM / 128, 1);
        gemm_bf16_wmma<<<grid, 256, 0, stream>>>(yb, Wpb, bp, bp, bp,
                                                 nullptr, nullptr, nullptr, out,
                                                 /*mode_base=*/3);
    }
}
